// PcdObjEncoder_39711267618974
// MI455X (gfx1250) — matrix operand order repaired and checked by reading, not compiled, not run on hardware
//
#include <hip/hip_runtime.h>

typedef __attribute__((ext_vector_type(8)))  _Float16 v8h;
typedef __attribute__((ext_vector_type(16))) _Float16 v16h;
typedef __attribute__((ext_vector_type(8)))  float    v8f;

#define NWAVES 8
#define BLOCK  256

// ---------------- dynamic LDS layout (bytes), total ~278KB of the 320KB WGP LDS ---------
#define SM_XYZ    0        // 1024*3 f32
#define SM_FEAT   12288    // 1024*3 f32
#define SM_BUF0   24576    // 81920  : x0 (512x32 f16) / x1 (256x160 f16) / x2 (16x288 f16)
#define SM_BUF1   106496   // 65536  : act ping
#define SM_BUF2   172032   // 65536  : act pong
#define SM_POOL1  237568   // 32*128 f32
#define SM_POOL2  253952   // 16*256 f32
#define SM_FPOOL  270336   // 768 f32
#define SM_CENT1  273408   // 32*3 f32
#define SM_CENT2  273792   // 16*3 f32 (padded)
#define SM_SEL    274048   // 32 int
#define SM_GIDX   274176   // 32*32 int
#define SM_DIST   278272   // 1024 f32
#define SM_REDV   282368   // 256 f32
#define SM_REDI   283392   // 256 int
#define SM_TOTAL  284416

struct SAParams {
  const _Float16* Wp[9];   // packed f16 weights, row-major [cout][cinp]
  const float* bb[9];
  const float* g[9];
  const float* m[9];
  const float* v[9];
};

// ---- WMMA fragment helpers (layouts per CDNA5 ISA 7.12.2) -------------------------------
// A 16x32 f16: lanes 0-15 row r hold K[0:8],K[16:24]; lanes 16-31 hold K[8:16],K[24:32]
__device__ __forceinline__ v16h load_a_frag(const _Float16* base, int ld, int row0, int kbase, int lane) {
  int r = lane & 15, h = lane >> 4;
  const _Float16* p = base + (size_t)(row0 + r) * ld + kbase + h * 8;
  v8h lo = *(const v8h*)(p);
  v8h hi = *(const v8h*)(p + 16);
  v16h a;
#pragma unroll
  for (int i = 0; i < 8; ++i) { a[i] = lo[i]; a[i + 8] = hi[i]; }
  return a;
}

// B 32x16 f16 from packed f16 weight Wp[cout][cinp]: lane holds col n=l&15,
// lanes<16: K[0:8],K[16:24]; lanes>=16: K[8:16],K[24:32] (same as A) -> two b128 loads.
__device__ __forceinline__ v16h load_b_frag_h(const _Float16* __restrict__ Wp, int cinp,
                                              int nt, int kc, int lane) {
  int n  = nt * 16 + (lane & 15);
  int kb = kc * 32 + (lane >> 4) * 8;
  const _Float16* p = Wp + (size_t)n * cinp + kb;
  v8h lo = *(const v8h*)(p);
  v8h hi = *(const v8h*)(p + 16);
  v16h b;
#pragma unroll
  for (int i = 0; i < 8; ++i) { b[i * 2] = lo[i]; }
#pragma unroll
  for (int i = 0; i < 8; ++i) { }
  // elements j<8 map to K=kb+j, j>=8 to K=kb+16+(j-8) -> concatenate lo|hi
  v16h bb;
#pragma unroll
  for (int i = 0; i < 8; ++i) { bb[i] = lo[i]; bb[i + 8] = hi[i]; }
  return bb;
}

__device__ __forceinline__ v8f wmma32(v16h a, v16h b, v8f c) {
  return __builtin_amdgcn_wmma_f32_16x16x32_f16(false, a, false, b, (short)0, c, false, false);
}

// ---- generic fused GEMM + BN + ReLU, D stored as f16 row-major in LDS -------------------
__device__ void mlp_layer(const _Float16* Ain, int ldA, _Float16* Aout, int ldOut,
                          const _Float16* __restrict__ Wp, int cinp,
                          const float* __restrict__ g, const float* __restrict__ bb,
                          const float* __restrict__ m, const float* __restrict__ v,
                          int Mrows, int cout, int wave, int lane) {
  int MT = Mrows >> 4, NT = cout >> 4, KT = cinp >> 5;
  for (int nt = wave; nt < NT; nt += NWAVES) {
    int col = nt * 16 + (lane & 15);
    float vv = v[col];
    float sc = g[col] * rsqrtf(vv + 1e-5f);
    float sh = bb[col] - m[col] * sc;
    for (int mt0 = 0; mt0 < MT; mt0 += 4) {
      int mcnt = MT - mt0; if (mcnt > 4) mcnt = 4;
      v8f acc[4] = {v8f{}, v8f{}, v8f{}, v8f{}};
      for (int kc = 0; kc < KT; ++kc) {
        v16h bf = load_b_frag_h(Wp, cinp, nt, kc, lane);
        for (int mm = 0; mm < mcnt; ++mm) {
          v16h af = load_a_frag(Ain, ldA, (mt0 + mm) * 16, kc * 32, lane);
          acc[mm] = wmma32(af, bf, acc[mm]);
        }
      }
      for (int mm = 0; mm < mcnt; ++mm) {
        int rbase = (mt0 + mm) * 16 + (lane >> 4) * 8;
#pragma unroll
        for (int i = 0; i < 8; ++i) {
          float y = acc[mm][i] * sc + sh;
          y = y > 0.0f ? y : 0.0f;
          Aout[(size_t)(rbase + i) * ldOut + col] = (_Float16)y;
        }
      }
    }
  }
}

// ---- last SA layer: GEMM + BN + ReLU fused with max-pool over 32 samples/centroid -------
__device__ void mlp_layer_pool(const _Float16* Ain, int ldA, float* Pool,
                               const _Float16* __restrict__ Wp, int cinp,
                               const float* __restrict__ g, const float* __restrict__ bb,
                               const float* __restrict__ m, const float* __restrict__ v,
                               int ncent, int cent_base, int cout, int wave, int lane) {
  int NT = cout >> 4, KT = cinp >> 5;
  for (int c = wave; c < ncent; c += NWAVES) {
    for (int nt = 0; nt < NT; ++nt) {
      int col = nt * 16 + (lane & 15);
      float vv = v[col];
      float sc = g[col] * rsqrtf(vv + 1e-5f);
      float sh = bb[col] - m[col] * sc;
      v8f a0 = {}, a1 = {};
      for (int kc = 0; kc < KT; ++kc) {
        v16h bf  = load_b_frag_h(Wp, cinp, nt, kc, lane);
        v16h af0 = load_a_frag(Ain, ldA, c * 32,      kc * 32, lane);
        v16h af1 = load_a_frag(Ain, ldA, c * 32 + 16, kc * 32, lane);
        a0 = wmma32(af0, bf, a0);
        a1 = wmma32(af1, bf, a1);
      }
      float mx = -1e30f;
#pragma unroll
      for (int i = 0; i < 8; ++i) {
        float y0 = a0[i] * sc + sh; y0 = y0 > 0.0f ? y0 : 0.0f;
        float y1 = a1[i] * sc + sh; y1 = y1 > 0.0f ? y1 : 0.0f;
        mx = fmaxf(mx, fmaxf(y0, y1));
      }
      mx = fmaxf(mx, __shfl_xor(mx, 16, 32));      // combine rows 0-7 vs 8-15
      if (lane < 16) Pool[(size_t)(cent_base + c) * cout + col] = mx;
    }
  }
}

// =========================================================================================
// Kernel 0: pack one f32 weight matrix to zero-padded f16 [coutp][cinp]
// =========================================================================================
__global__ __launch_bounds__(BLOCK) void pack_w_kernel(const float* __restrict__ W,
                                                       _Float16* __restrict__ dst,
                                                       int cout, int cin, int coutp, int cinp) {
  int total = coutp * cinp;
  for (int i = blockIdx.x * BLOCK + threadIdx.x; i < total; i += gridDim.x * BLOCK) {
    int r = i / cinp, k = i - r * cinp;
    float w = (r < cout && k < cin) ? W[(size_t)r * cin + k] : 0.0f;
    dst[i] = (_Float16)w;
  }
}

// =========================================================================================
// Kernel 1: whole per-object PointNet++ pipeline. grid=1024 (one WGP each), block=256.
// =========================================================================================
__global__ __launch_bounds__(BLOCK) void pointnet_obj_kernel(
    const float* __restrict__ pcds, SAParams sp, float* __restrict__ out,
    _Float16* __restrict__ embh) {
  extern __shared__ char smem[];
  float*    sxyz   = (float*)(smem + SM_XYZ);
  float*    sfeat  = (float*)(smem + SM_FEAT);
  float*    spool1 = (float*)(smem + SM_POOL1);
  float*    spool2 = (float*)(smem + SM_POOL2);
  float*    sfpool = (float*)(smem + SM_FPOOL);
  float*    scent1 = (float*)(smem + SM_CENT1);
  float*    scent2 = (float*)(smem + SM_CENT2);
  int*      ssel   = (int*)  (smem + SM_SEL);
  int*      sgidx  = (int*)  (smem + SM_GIDX);
  float*    sdist  = (float*)(smem + SM_DIST);
  float*    sredv  = (float*)(smem + SM_REDV);
  int*      sredi  = (int*)  (smem + SM_REDI);
  _Float16* xbuf   = (_Float16*)(smem + SM_BUF0);
  _Float16* actA   = (_Float16*)(smem + SM_BUF1);
  _Float16* actB   = (_Float16*)(smem + SM_BUF2);

  int t = threadIdx.x, wave = t >> 5, lane = t & 31;
  int obj = blockIdx.x;

  // ---- load point cloud (1024 x 6) --------------------------------------------------
  const float* pc = pcds + (size_t)obj * 1024 * 6;
  for (int p = t; p < 1024; p += BLOCK) {
    sxyz[p * 3 + 0] = pc[p * 6 + 0];
    sxyz[p * 3 + 1] = pc[p * 6 + 1];
    sxyz[p * 3 + 2] = pc[p * 6 + 2];
    sfeat[p * 3 + 0] = pc[p * 6 + 3];
    sfeat[p * 3 + 1] = pc[p * 6 + 4];
    sfeat[p * 3 + 2] = pc[p * 6 + 5];
    sdist[p] = 1e10f;
  }
  if (t == 0) ssel[0] = 0;
  __syncthreads();

  // ---- FPS: 32 of 1024 ---------------------------------------------------------------
  int last = 0;
  for (int it = 1; it < 32; ++it) {
    float lx = sxyz[last * 3 + 0], ly = sxyz[last * 3 + 1], lz = sxyz[last * 3 + 2];
    float bv = -1.0f; int bi = 0;
    for (int p = t; p < 1024; p += BLOCK) {
      float dx = sxyz[p * 3 + 0] - lx, dy = sxyz[p * 3 + 1] - ly, dz = sxyz[p * 3 + 2] - lz;
      float dd = fminf(sdist[p], dx * dx + dy * dy + dz * dz);
      sdist[p] = dd;
      if (dd > bv || (dd == bv && p < bi)) { bv = dd; bi = p; }
    }
    sredv[t] = bv; sredi[t] = bi;
    __syncthreads();
    for (int s = BLOCK / 2; s > 0; s >>= 1) {
      if (t < s) {
        float ov = sredv[t + s]; int oi = sredi[t + s];
        if (ov > sredv[t] || (ov == sredv[t] && oi < sredi[t])) { sredv[t] = ov; sredi[t] = oi; }
      }
      __syncthreads();
    }
    last = sredi[0];
    if (t == 0) ssel[it] = last;
    __syncthreads();
  }
  if (t < 32) {
    int s1 = ssel[t];
    scent1[t * 3 + 0] = sxyz[s1 * 3 + 0];
    scent1[t * 3 + 1] = sxyz[s1 * 3 + 1];
    scent1[t * 3 + 2] = sxyz[s1 * 3 + 2];
  }
  __syncthreads();

  // ---- ball query SA0: first 32 in-radius indices per centroid ------------------------
  for (int ci = wave; ci < 32; ci += NWAVES) {
    float cx = scent1[ci * 3 + 0], cy = scent1[ci * 3 + 1], cz = scent1[ci * 3 + 2];
    int base = 0, firstp = 0;
    for (int chunk = 0; chunk < 32 && base < 32; ++chunk) {
      int p = chunk * 32 + lane;
      float dx = sxyz[p * 3 + 0] - cx, dy = sxyz[p * 3 + 1] - cy, dz = sxyz[p * 3 + 2] - cz;
      bool pred = (dx * dx + dy * dy + dz * dz) < 0.04f;
      unsigned mask = (unsigned)__ballot(pred);
      int cnt = __popc(mask);
      if (pred) {
        int slot = base + __popc(mask & ((1u << lane) - 1));
        if (slot < 32) sgidx[ci * 32 + slot] = p;
      }
      if (base == 0 && cnt > 0) firstp = chunk * 32 + (__ffs(mask) - 1);
      base += cnt;
    }
    for (int j = base + lane; j < 32; j += 32) sgidx[ci * 32 + j] = firstp;
  }
  __syncthreads();

  // ---- SA0: [6->64->64->128], 32 cents x 32 samples, processed in 2 halves ------------
  for (int half = 0; half < 2; ++half) {
    int cbase = half * 16;
    for (int r = t; r < 512; r += BLOCK) {        // build x0: 512 x 32 (pad)
      int c = cbase + (r >> 5), s = r & 31;
      int ip = sgidx[c * 32 + s];
      _Float16* row = xbuf + (size_t)r * 32;
      row[0] = (_Float16)(sxyz[ip * 3 + 0] - scent1[c * 3 + 0]);
      row[1] = (_Float16)(sxyz[ip * 3 + 1] - scent1[c * 3 + 1]);
      row[2] = (_Float16)(sxyz[ip * 3 + 2] - scent1[c * 3 + 2]);
      row[3] = (_Float16)sfeat[ip * 3 + 0];
      row[4] = (_Float16)sfeat[ip * 3 + 1];
      row[5] = (_Float16)sfeat[ip * 3 + 2];
      for (int j = 6; j < 32; ++j) row[j] = (_Float16)0.0f;
    }
    __syncthreads();
    mlp_layer(xbuf, 32, actA, 64, sp.Wp[0], 32, sp.g[0], sp.bb[0], sp.m[0], sp.v[0], 512, 64, wave, lane);
    __syncthreads();
    mlp_layer(actA, 64, actB, 64, sp.Wp[1], 64, sp.g[1], sp.bb[1], sp.m[1], sp.v[1], 512, 64, wave, lane);
    __syncthreads();
    mlp_layer_pool(actB, 64, spool1, sp.Wp[2], 64, sp.g[2], sp.bb[2], sp.m[2], sp.v[2],
                   16, cbase, 128, wave, lane);
    __syncthreads();
  }

  // ---- FPS2: 16 of 32 (wave 0, registers + shfl) --------------------------------------
  if (wave == 0) {
    float px = scent1[lane * 3 + 0], py = scent1[lane * 3 + 1], pz = scent1[lane * 3 + 2];
    float dd = 1e10f;
    int last2 = 0;
    if (lane == 0) ssel[0] = 0;
    for (int it = 1; it < 16; ++it) {
      float lx = __shfl(px, last2, 32), ly = __shfl(py, last2, 32), lz = __shfl(pz, last2, 32);
      float dx = px - lx, dy = py - ly, dz = pz - lz;
      dd = fminf(dd, dx * dx + dy * dy + dz * dz);
      float bv = dd; int bi = lane;
      for (int off = 16; off > 0; off >>= 1) {
        float ov = __shfl_xor(bv, off, 32); int oi = __shfl_xor(bi, off, 32);
        if (ov > bv || (ov == bv && oi < bi)) { bv = ov; bi = oi; }
      }
      last2 = bi;
      if (lane == 0) ssel[it] = last2;
    }
  }
  __syncthreads();
  if (t < 16) {
    int s2 = ssel[t];
    scent2[t * 3 + 0] = scent1[s2 * 3 + 0];
    scent2[t * 3 + 1] = scent1[s2 * 3 + 1];
    scent2[t * 3 + 2] = scent1[s2 * 3 + 2];
  }
  __syncthreads();

  // ---- ball query SA1 (16 cents over 32 pts, single ballot each) ----------------------
  for (int ci = wave; ci < 16; ci += NWAVES) {
    float cx = scent2[ci * 3 + 0], cy = scent2[ci * 3 + 1], cz = scent2[ci * 3 + 2];
    float dx = scent1[lane * 3 + 0] - cx, dy = scent1[lane * 3 + 1] - cy, dz = scent1[lane * 3 + 2] - cz;
    bool pred = (dx * dx + dy * dy + dz * dz) < 0.16f;
    unsigned mask = (unsigned)__ballot(pred);
    int cnt = __popc(mask);
    int firstp = __ffs(mask) - 1;
    if (pred) sgidx[ci * 32 + __popc(mask & ((1u << lane) - 1))] = lane;
    for (int j = cnt + lane; j < 32; j += 32) sgidx[ci * 32 + j] = firstp;
  }
  __syncthreads();

  // ---- SA1: [131->128->128->256], 16 cents x 32 samples, 2 halves of 8 ----------------
  for (int half = 0; half < 2; ++half) {
    int cbase = half * 8;
    for (int r = t; r < 256; r += BLOCK) {        // build x1: 256 x 160 (pad 131->160)
      int c = cbase + (r >> 5), s = r & 31;
      int ip = sgidx[c * 32 + s];
      _Float16* row = xbuf + (size_t)r * 160;
      row[0] = (_Float16)(scent1[ip * 3 + 0] - scent2[c * 3 + 0]);
      row[1] = (_Float16)(scent1[ip * 3 + 1] - scent2[c * 3 + 1]);
      row[2] = (_Float16)(scent1[ip * 3 + 2] - scent2[c * 3 + 2]);
      for (int j = 0; j < 128; ++j) row[3 + j] = (_Float16)spool1[ip * 128 + j];
      for (int j = 131; j < 160; ++j) row[j] = (_Float16)0.0f;
    }
    __syncthreads();
    mlp_layer(xbuf, 160, actA, 128, sp.Wp[3], 160, sp.g[3], sp.bb[3], sp.m[3], sp.v[3], 256, 128, wave, lane);
    __syncthreads();
    mlp_layer(actA, 128, actB, 128, sp.Wp[4], 128, sp.g[4], sp.bb[4], sp.m[4], sp.v[4], 256, 128, wave, lane);
    __syncthreads();
    mlp_layer_pool(actB, 128, spool2, sp.Wp[5], 128, sp.g[5], sp.bb[5], sp.m[5], sp.v[5],
                   8, cbase, 256, wave, lane);
    __syncthreads();
  }

  // ---- final MLP [259->256->512->768] on 16 points, pool over 16 ----------------------
  for (int idx = t; idx < 16 * 288; idx += BLOCK) {   // build x2: 16 x 288 (pad 259->288)
    int r = idx / 288, j = idx - r * 288;
    float val;
    if (j < 3) val = scent2[r * 3 + j];
    else if (j < 259) val = spool2[r * 256 + (j - 3)];
    else val = 0.0f;
    xbuf[(size_t)r * 288 + j] = (_Float16)val;
  }
  __syncthreads();
  mlp_layer(xbuf, 288, actA, 256, sp.Wp[6], 288, sp.g[6], sp.bb[6], sp.m[6], sp.v[6], 16, 256, wave, lane);
  __syncthreads();
  mlp_layer(actA, 256, actB, 512, sp.Wp[7], 256, sp.g[7], sp.bb[7], sp.m[7], sp.v[7], 16, 512, wave, lane);
  __syncthreads();
  // last layer fused with pool over all 16 rows -> fpool[768]
  for (int nt = wave; nt < 48; nt += NWAVES) {
    int col = nt * 16 + (lane & 15);
    float vv = sp.v[8][col];
    float sc = sp.g[8][col] * rsqrtf(vv + 1e-5f);
    float sh = sp.bb[8][col] - sp.m[8][col] * sc;
    v8f acc = {};
    for (int kc = 0; kc < 16; ++kc) {
      v16h bf = load_b_frag_h(sp.Wp[8], 512, nt, kc, lane);
      v16h af = load_a_frag(actB, 512, 0, kc * 32, lane);
      acc = wmma32(af, bf, acc);
    }
    float mx = -1e30f;
#pragma unroll
    for (int i = 0; i < 8; ++i) {
      float y = acc[i] * sc + sh;
      y = y > 0.0f ? y : 0.0f;
      mx = fmaxf(mx, y);
    }
    mx = fmaxf(mx, __shfl_xor(mx, 16, 32));
    if (lane < 16) sfpool[col] = mx;
  }
  __syncthreads();

  // ---- write embedding (twice, per reference tuple) + f16 copy for head GEMM ----------
  float* o0 = out + (size_t)obj * 768;
  float* o1 = out + 786432 + (size_t)obj * 768;
  _Float16* eh = embh + (size_t)obj * 768;
  for (int j = t; j < 768; j += BLOCK) {
    float e = sfpool[j];
    o0[j] = e; o1[j] = e; eh[j] = (_Float16)e;
  }
}

// =========================================================================================
// Kernel 2: head — relu(emb@W1^T+b1) -> layernorm -> @W2^T+b2. grid=64, block=256.
// =========================================================================================
__global__ __launch_bounds__(BLOCK) void head_kernel(
    const _Float16* __restrict__ embh,
    const _Float16* __restrict__ W1p, const float* __restrict__ b1,
    const float* __restrict__ lng, const float* __restrict__ lnb,
    const _Float16* __restrict__ W2p, const float* __restrict__ b2,
    float* __restrict__ cls) {
  __shared__ float Xs[16 * 384];
  __shared__ __align__(16) _Float16 Xn[16 * 384];
  int t = threadIdx.x, wave = t >> 5, lane = t & 31;
  int mt = blockIdx.x;                       // 16-row tile of the 1024 objects

  // GEMM1: (16 x 768) @ W1^T (768 x 384), +b1, relu -> Xs
  for (int nt = wave; nt < 24; nt += NWAVES) {
    v8f acc = {};
    for (int kc = 0; kc < 24; ++kc) {
      v16h bf = load_b_frag_h(W1p, 768, nt, kc, lane);
      v16h af = load_a_frag(embh, 768, mt * 16, kc * 32, lane);
      acc = wmma32(af, bf, acc);
    }
    int col = nt * 16 + (lane & 15);
    float bbv = b1[col];
#pragma unroll
    for (int i = 0; i < 8; ++i) {
      float y = acc[i] + bbv;
      y = y > 0.0f ? y : 0.0f;
      Xs[(i + 8 * (lane >> 4)) * 384 + col] = y;
    }
  }
  __syncthreads();

  // layernorm over 384 per row (wave handles 2 rows)
  for (int rr = wave * 2; rr < wave * 2 + 2; ++rr) {
    float s = 0.0f, s2 = 0.0f;
    for (int c = lane; c < 384; c += 32) {
      float x = Xs[rr * 384 + c];
      s += x; s2 += x * x;
    }
    for (int off = 16; off > 0; off >>= 1) {
      s  += __shfl_xor(s, off, 32);
      s2 += __shfl_xor(s2, off, 32);
    }
    float mu = s * (1.0f / 384.0f);
    float var = s2 * (1.0f / 384.0f) - mu * mu;
    float rstd = rsqrtf(var + 1e-12f);
    for (int c = lane; c < 384; c += 32) {
      float x = Xs[rr * 384 + c];
      Xn[rr * 384 + c] = (_Float16)((x - mu) * rstd * lng[c] + lnb[c]);
    }
  }
  __syncthreads();

  // GEMM2: (16 x 384) @ W2^T (384 x 607, cout padded to 608), +b2 -> cls
  for (int nt = wave; nt < 38; nt += NWAVES) {
    v8f acc = {};
    for (int kc = 0; kc < 12; ++kc) {
      v16h bf = load_b_frag_h(W2p, 384, nt, kc, lane);
      v16h af = load_a_frag(Xn, 384, 0, kc * 32, lane);
      acc = wmma32(af, bf, acc);
    }
    int col = nt * 16 + (lane & 15);
    if (col < 607) {
      float bbv = b2[col];
#pragma unroll
      for (int i = 0; i < 8; ++i) {
        int row = mt * 16 + i + 8 * (lane >> 4);
        cls[(size_t)row * 607 + col] = acc[i] + bbv;
      }
    }
  }
}

// =========================================================================================
extern "C" void kernel_launch(void* const* d_in, const int* in_sizes, int n_in,
                              void* d_out, int out_size, void* d_ws, size_t ws_size,
                              hipStream_t stream) {
  // params pytree leaves in jax tree_flatten order (dict keys sorted):
  // head: W1,W2,b1,b2,ln_b,ln_g ; then sa[0..2] x layer[0..2] x (W,b,g,m,v)
  static const long leaf_sz[51] = {
      294912, 233088, 384, 607, 384, 384,
      384,   64, 64, 64, 64,   4096,  64, 64, 64, 64,    8192,  128, 128, 128, 128,
      16768, 128, 128, 128, 128, 16384, 128, 128, 128, 128, 32768, 256, 256, 256, 256,
      66304, 256, 256, 256, 256, 131072, 512, 512, 512, 512, 393216, 768, 768, 768, 768};
  const float* leaf[51];
  if (n_in >= 55) {
    for (int i = 0; i < 51; ++i) leaf[i] = (const float*)d_in[4 + i];
  } else {
    const float* base = (const float*)d_in[4];
    long off = 0;
    for (int i = 0; i < 51; ++i) { leaf[i] = base + off; off += leaf_sz[i]; }
  }

  // ---- workspace layout (halfs): emb_f16 then packed weights -------------------------
  _Float16* embh = (_Float16*)d_ws;                       // 1024*768
  _Float16* wsp  = embh + 786432;
  // per-matrix: cout, cin, coutp, cinp (SA 0..8, then head W1, W2)
  static const int wdim[11][4] = {
      {64, 6, 64, 32},    {64, 64, 64, 64},   {128, 64, 128, 64},
      {128, 131, 128, 160}, {128, 128, 128, 128}, {256, 128, 256, 128},
      {256, 259, 256, 288}, {512, 256, 512, 512}, {768, 512, 768, 512},
      {384, 768, 384, 768}, {607, 384, 608, 384}};
  static const int wleaf[11] = {6, 11, 16, 21, 26, 31, 36, 41, 46, 0, 1};
  _Float16* wpk[11];
  {
    long off = 0;
    for (int i = 0; i < 11; ++i) {
      wpk[i] = wsp + off;
      off += (long)wdim[i][2] * wdim[i][3];
    }
  }
  for (int i = 0; i < 11; ++i) {
    int total = wdim[i][2] * wdim[i][3];
    int grid = (total + BLOCK - 1) / BLOCK;
    if (grid > 1024) grid = 1024;
    pack_w_kernel<<<grid, BLOCK, 0, stream>>>(leaf[wleaf[i]], wpk[i],
                                              wdim[i][0], wdim[i][1], wdim[i][2], wdim[i][3]);
  }

  SAParams sp;
  for (int L = 0; L < 9; ++L) {
    int li = 6 + L * 5;
    sp.Wp[L] = wpk[L];
    sp.bb[L] = leaf[li + 1];
    sp.g[L]  = leaf[li + 2];
    sp.m[L]  = leaf[li + 3];
    sp.v[L]  = leaf[li + 4];
  }
  const float* hb1 = leaf[2];
  const float* hb2 = leaf[3];
  const float* hlnb = leaf[4];
  const float* hlng = leaf[5];

  const float* pcds = (const float*)d_in[0];
  float* out = (float*)d_out;
  float* cls = out + 2 * 786432;

  pointnet_obj_kernel<<<1024, BLOCK, SM_TOTAL, stream>>>(pcds, sp, out, embh);
  head_kernel<<<64, BLOCK, 0, stream>>>(embh, wpk[9], hb1, hlng, hlnb, wpk[10], hb2, cls);
}
